// Gemma2Attention_65103114272907
// MI455X (gfx1250) — matrix operand order repaired and checked by reading, not compiled, not run on hardware
//
#include <hip/hip_runtime.h>

typedef __attribute__((ext_vector_type(16))) _Float16 v16h;
typedef __attribute__((ext_vector_type(8)))  _Float16 v8h;
typedef __attribute__((ext_vector_type(2)))  _Float16 v2h;
typedef __attribute__((ext_vector_type(8)))  float    v8f;

// ---- fragment helpers -------------------------------------------------------
// Build a v16h A/B fragment from two 16B-aligned 8-half chunks (ds_read_b128 x2)
__device__ __forceinline__ v16h frag16(const _Float16* p0, const _Float16* p1) {
  v8h lo = *(const v8h*)p0;
  v8h hi = *(const v8h*)p1;
  v16h r;
#pragma unroll
  for (int i = 0; i < 8; ++i) { r[i] = lo[i]; r[i + 8] = hi[i]; }
  return r;
}

__device__ __forceinline__ v8f wmma_f16(v16h a, v16h b, v8f c) {
  // (neg_a, A, neg_b, B, c_mod, C, reuse_a, reuse_b)
  return __builtin_amdgcn_wmma_f32_16x16x32_f16(false, a, false, b, (short)0, c,
                                                false, false);
}

// Stage 16 contiguous elements into LDS as f16 (fp32 source: convert; f16: copy)
__device__ __forceinline__ void load16_to_lds(const float* src, _Float16* dst) {
  const float4* p = (const float4*)src;
  float4 f0 = p[0], f1 = p[1], f2 = p[2], f3 = p[3];
  v8h h0, h1;
  h0[0]=(_Float16)f0.x; h0[1]=(_Float16)f0.y; h0[2]=(_Float16)f0.z; h0[3]=(_Float16)f0.w;
  h0[4]=(_Float16)f1.x; h0[5]=(_Float16)f1.y; h0[6]=(_Float16)f1.z; h0[7]=(_Float16)f1.w;
  h1[0]=(_Float16)f2.x; h1[1]=(_Float16)f2.y; h1[2]=(_Float16)f2.z; h1[3]=(_Float16)f2.w;
  h1[4]=(_Float16)f3.x; h1[5]=(_Float16)f3.y; h1[6]=(_Float16)f3.z; h1[7]=(_Float16)f3.w;
  *(v8h*)dst = h0; *(v8h*)(dst + 8) = h1;
}
__device__ __forceinline__ void load16_to_lds(const _Float16* src, _Float16* dst) {
  const v8h* p = (const v8h*)src;
  *(v8h*)dst = p[0]; *(v8h*)(dst + 8) = p[1];
}

// ---- generic WMMA GEMM: C(MxN) = A(MxK) * B(KxN), A f32/f16, B f32 weights ---
// 256 threads = 8 waves; block tile 128x128, k-step 32; wave tile 32x64.
// Double-buffered LDS: stage k-step i+1 while WMMAs consume k-step i.
constexpr int LDT = 40;  // padded k-stride in halves (16B-aligned frags, no bank conflicts)

template <typename TA, typename TC>
__global__ __launch_bounds__(256) void gemm_wmma(const TA* __restrict__ A,
                                                 const float* __restrict__ Bw,
                                                 TC* __restrict__ C,
                                                 int M, int N, int K) {
  __shared__ _Float16 Alds[2][128 * LDT];   // [m][k] k-contiguous
  __shared__ _Float16 Blds[2][128 * LDT];   // [n][k] k-contiguous (transposed at stage)
  const int tid  = threadIdx.x;
  const int lane = tid & 31;
  const int wave = tid >> 5;
  const int wm   = wave >> 1;   // 0..3
  const int wn   = wave & 1;    // 0..1
  const int g    = lane >> 4;   // lane group (WMMA fragment layout)
  const int l16  = lane & 15;
  const int bm   = blockIdx.y * 128;
  const int bn   = blockIdx.x * 128;

  const int arow = tid >> 1;          // 0..127
  const int ah   = (tid & 1) * 16;    // 0/16
  const int bcol = tid >> 1;
  const int bkh  = (tid & 1) * 16;

  auto stage = [&](int k0, int buf) {
    // A tile (128x32), 16 elems/thread, b128 global loads
    load16_to_lds(A + (size_t)(bm + arow) * K + k0 + ah,
                  &Alds[buf][arow * LDT + ah]);
    // B tile transposed: Blds[n][k] = Bw[k0+k][bn+n] (coalesced across threads)
    const float* src = Bw + (size_t)(k0 + bkh) * N + bn + bcol;
    _Float16* dst = &Blds[buf][bcol * LDT + bkh];
#pragma unroll
    for (int i = 0; i < 16; i += 2) {
      v2h t;
      t[0] = (_Float16)src[(size_t)i * N];
      t[1] = (_Float16)src[(size_t)(i + 1) * N];
      *(v2h*)(dst + i) = t;
    }
  };

  v8f acc[2][4];
#pragma unroll
  for (int i = 0; i < 2; ++i)
#pragma unroll
    for (int j = 0; j < 4; ++j) acc[i][j] = {};

  stage(0, 0);
  for (int k0 = 0; k0 < K; k0 += 32) {
    const int cur = (k0 >> 5) & 1;
    __syncthreads();                       // stage(cur) complete; buf cur^1 free (WAR)
    if (k0 + 32 < K) stage(k0 + 32, cur ^ 1);  // prefetch next tile under compute
#pragma unroll
    for (int mt = 0; mt < 2; ++mt) {
      const _Float16* ap = &Alds[cur][(wm * 32 + mt * 16 + l16) * LDT];
      v16h af = frag16(ap + g * 8, ap + 16 + g * 8);
#pragma unroll
      for (int nt = 0; nt < 4; ++nt) {
        const _Float16* bp = &Blds[cur][(wn * 64 + nt * 16 + l16) * LDT];
        v16h bf = frag16(bp + g * 8, bp + 16 + g * 8);
        acc[mt][nt] = wmma_f16(af, bf, acc[mt][nt]);
      }
    }
  }
#pragma unroll
  for (int mt = 0; mt < 2; ++mt)
#pragma unroll
    for (int nt = 0; nt < 4; ++nt) {
      const int row0 = bm + wm * 32 + mt * 16 + g * 8;
      const int col  = bn + wn * 64 + nt * 16 + l16;
#pragma unroll
      for (int r = 0; r < 8; ++r)
        C[(size_t)(row0 + r) * N + col] = (TC)acc[mt][nt][r];
    }
}

// ---- RoPE (in-place on f16 q/k buffers) -------------------------------------
__global__ void rope_kernel(_Float16* __restrict__ buf,
                            const float* __restrict__ cosb,
                            const float* __restrict__ sinb,
                            int nheads, int total) {
  int idx = blockIdx.x * blockDim.x + threadIdx.x;
  if (idx >= total) return;
  const int d = idx & 63;
  const int h = (idx >> 6) % nheads;
  const int t = idx / (64 * nheads);           // b*S + s
  _Float16* p = buf + ((size_t)t * nheads + h) * 128;
  const float x0 = (float)p[d];
  const float x1 = (float)p[d + 64];
  const float c0 = cosb[(size_t)t * 128 + d];
  const float s0 = sinb[(size_t)t * 128 + d];
  const float c1 = cosb[(size_t)t * 128 + d + 64];
  const float s1 = sinb[(size_t)t * 128 + d + 64];
  p[d]      = (_Float16)(x0 * c0 - x1 * s0);   // rotate_half: -x[d+64] for d<64
  p[d + 64] = (_Float16)(x1 * c1 + x0 * s1);   //              +x[d-64] for d>=64
}

// ---- flash attention with WMMA, sliding-window causal mask + tanh softcap ---
// grid = B*NH*(S/128); block = 256 (8 waves); wave owns a 16-query strip.
// K/V tiles double-buffered in LDS; compute skipped (wave-uniform) for key
// blocks entirely outside a strip's causal/sliding window.
__global__ __launch_bounds__(256) void attn_kernel(const _Float16* __restrict__ q,
                                                   const _Float16* __restrict__ k,
                                                   const _Float16* __restrict__ v,
                                                   _Float16* __restrict__ o) {
  constexpr int S = 2048, NH = 16, NKV = 8, HD = 128, WIN = 1024;
  constexpr int KLD = 136, VLD = 40, PLD = 40;
  const float pre = 0.0625f / 50.0f;          // SCALE / SOFTCAP
  __shared__ _Float16 Klds[2][32 * KLD];      // [key][hd], hd-contiguous
  __shared__ _Float16 Vlds[2][128 * VLD];     // [hd][key], key-contiguous
  __shared__ _Float16 Plds[8 * 16 * PLD];     // per-wave P relayout scratch

  const int tid = threadIdx.x, lane = tid & 31, wave = tid >> 5;
  const int g = lane >> 4, l16 = lane & 15;
  const int blk = blockIdx.x;
  const int qb = blk & 15;                    // S/128 = 16
  const int h  = (blk >> 4) & 15;             // NH
  const int b  = blk >> 8;
  const int q0 = qb * 128;
  const int kvh = h >> 1;                     // GQA: NH/NKV = 2

  auto stageKV = [&](int kb, int buf) {
    // K tile 32x128, direct f16 copy via b128 loads/stores
    const int kl = tid >> 3, seg = (tid & 7) * 16;
    load16_to_lds(k + (((size_t)b * S + kb + kl) * NKV + kvh) * HD + seg,
                  &Klds[buf][kl * KLD + seg]);
    // V tile transposed: Vlds[hd][key] (reads coalesce across threads per row)
    const int n = tid >> 1, kh = (tid & 1) * 16;
    const _Float16* src = v + (((size_t)b * S + kb + kh) * NKV + kvh) * HD + n;
    _Float16* dst = &Vlds[buf][n * VLD + kh];
#pragma unroll
    for (int i = 0; i < 16; i += 2) {
      v2h t;
      t[0] = src[(size_t)i * (NKV * HD)];
      t[1] = src[(size_t)(i + 1) * (NKV * HD)];
      *(v2h*)(dst + i) = t;
    }
  };

  // Q fragments (16 queries x 128 hd), loaded once, live in registers
  const int qrow = q0 + wave * 16 + l16;
  const _Float16* qp = q + (((size_t)b * S + qrow) * NH + h) * HD;
  v16h qf[4];
#pragma unroll
  for (int st = 0; st < 4; ++st)
    qf[st] = frag16(qp + st * 32 + g * 8, qp + st * 32 + 16 + g * 8);

  v8f acc[8];
#pragma unroll
  for (int i = 0; i < 8; ++i) acc[i] = {};
  float m8[8], l8[8];
#pragma unroll
  for (int r = 0; r < 8; ++r) { m8[r] = -3.0e38f; l8[r] = 0.0f; }

  const int kstart = (q0 >= WIN) ? (q0 - WIN) : 0;   // multiple of 32
  const int kend = q0 + 128;
  const int qlo = q0 + wave * 16;                    // wave's query strip
  const int qhi = qlo + 15;

  stageKV(kstart, 0);
  int it = 0;
  for (int kb = kstart; kb < kend; kb += 32, ++it) {
    const int cur = it & 1;
    __syncthreads();                            // stage(cur) done; buf cur^1 free
    if (kb + 32 < kend) stageKV(kb + 32, cur ^ 1);

    // wave-uniform skip: block entirely above strip (future) or below its window
    if (kb > qhi || (kb + 31) < (qlo - (WIN - 1))) continue;

    // S = Q K^T : two 16x16 tiles (keys kb..kb+15, kb+16..kb+31), K-dim = HD
    v8f s0 = {}, s1 = {};
#pragma unroll
    for (int st = 0; st < 4; ++st) {
      const _Float16* kp0 = &Klds[cur][l16 * KLD + st * 32];
      s0 = wmma_f16(qf[st], frag16(kp0 + g * 8, kp0 + 16 + g * 8), s0);
      const _Float16* kp1 = &Klds[cur][(16 + l16) * KLD + st * 32];
      s1 = wmma_f16(qf[st], frag16(kp1 + g * 8, kp1 + 16 + g * 8), s1);
    }

    // softcap + analytic mask + online softmax (row = r + 8*g, col = l16)
    float p0[8], p1[8];
#pragma unroll
    for (int r = 0; r < 8; ++r) {
      const int qi  = qlo + r + 8 * g;
      const int k0i = kb + l16;
      const int k1i = k0i + 16;
      float x0 = tanhf(s0[r] * pre) * 50.0f;
      float x1 = tanhf(s1[r] * pre) * 50.0f;
      if (k0i > qi || (qi - k0i) >= WIN) x0 = -1.0e9f;
      if (k1i > qi || (qi - k1i) >= WIN) x1 = -1.0e9f;
      float mx = fmaxf(x0, x1);
#pragma unroll
      for (int off = 1; off < 16; off <<= 1)
        mx = fmaxf(mx, __shfl_xor(mx, off, 32));
      const float mnew = fmaxf(m8[r], mx);
      const float f = __expf(m8[r] - mnew);   // fully-masked blocks self-correct later
      m8[r] = mnew;
      x0 = __expf(x0 - mnew);
      x1 = __expf(x1 - mnew);
      float sum = x0 + x1;
#pragma unroll
      for (int off = 1; off < 16; off <<= 1)
        sum += __shfl_xor(sum, off, 32);
      l8[r] = l8[r] * f + sum;
      p0[r] = x0; p1[r] = x1;
#pragma unroll
      for (int nt = 0; nt < 8; ++nt) acc[nt][r] = acc[nt][r] * f;
    }

    // D-fragment -> A-fragment relayout of P via per-wave LDS
    _Float16* pl = &Plds[wave * 16 * PLD];
#pragma unroll
    for (int r = 0; r < 8; ++r) {
      const int row = r + 8 * g;
      pl[row * PLD + l16]      = (_Float16)p0[r];
      pl[row * PLD + 16 + l16] = (_Float16)p1[r];
    }
    asm volatile("s_wait_dscnt 0" ::: "memory");  // same-wave LDS RAW fence
    v16h pf = frag16(&pl[l16 * PLD + g * 8], &pl[l16 * PLD + 16 + g * 8]);

    // O += P V : 8 N-tiles over HD, K-dim = 32 keys
#pragma unroll
    for (int nt = 0; nt < 8; ++nt) {
      const _Float16* vp = &Vlds[cur][(nt * 16 + l16) * VLD];
      acc[nt] = wmma_f16(pf, frag16(vp + g * 8, vp + 16 + g * 8), acc[nt]);
    }
  }

  // normalize and store (layout matches (B*S) x (NH*HD) row-major for O-proj)
  _Float16* op = o + (((size_t)b * S + q0 + wave * 16) * NH + h) * HD;
#pragma unroll
  for (int nt = 0; nt < 8; ++nt)
#pragma unroll
    for (int r = 0; r < 8; ++r) {
      const int row = r + 8 * g;
      op[(size_t)row * (NH * HD) + nt * 16 + l16] = (_Float16)(acc[nt][r] / l8[r]);
    }
}

// ---- host launcher ----------------------------------------------------------
extern "C" void kernel_launch(void* const* d_in, const int* in_sizes, int n_in,
                              void* d_out, int out_size, void* d_ws, size_t ws_size,
                              hipStream_t stream) {
  (void)in_sizes; (void)n_in; (void)out_size; (void)ws_size;
  const float* x    = (const float*)d_in[0];
  const float* wq   = (const float*)d_in[1];
  const float* wk   = (const float*)d_in[2];
  const float* wv   = (const float*)d_in[3];
  const float* wo   = (const float*)d_in[4];
  const float* cosb = (const float*)d_in[5];
  const float* sinb = (const float*)d_in[6];
  // d_in[7] = attention_mask: not read; causal+window mask computed analytically.
  float* out = (float*)d_out;

  constexpr int Btot = 2, S = 2048, HID = 2048, NH = 16, NKV = 8, HD = 128;
  constexpr size_t M = (size_t)Btot * S;  // 4096 rows

  // workspace (f16): q(16MB) + k(8MB) + v(8MB) + o(16MB) ~= 50 MB
  _Float16* qbuf = (_Float16*)d_ws;
  _Float16* kbuf = qbuf + M * NH * HD;
  _Float16* vbuf = kbuf + M * NKV * HD;
  _Float16* obuf = vbuf + M * NKV * HD;

  dim3 blk(256);
  gemm_wmma<float, _Float16><<<dim3(NH * HD / 128, M / 128), blk, 0, stream>>>(
      x, wq, qbuf, (int)M, NH * HD, HID);
  gemm_wmma<float, _Float16><<<dim3(NKV * HD / 128, M / 128), blk, 0, stream>>>(
      x, wk, kbuf, (int)M, NKV * HD, HID);
  gemm_wmma<float, _Float16><<<dim3(NKV * HD / 128, M / 128), blk, 0, stream>>>(
      x, wv, vbuf, (int)M, NKV * HD, HID);

  const int totq = (int)(M * NH * 64);
  rope_kernel<<<(totq + 255) / 256, blk, 0, stream>>>(qbuf, cosb, sinb, NH, totq);
  const int totk = (int)(M * NKV * 64);
  rope_kernel<<<(totk + 255) / 256, blk, 0, stream>>>(kbuf, cosb, sinb, NKV, totk);

  attn_kernel<<<dim3(Btot * NH * (S / 128)), blk, 0, stream>>>(qbuf, kbuf, vbuf, obuf);

  gemm_wmma<_Float16, float><<<dim3(HID / 128, M / 128), blk, 0, stream>>>(
      obuf, wo, out, (int)M, HID, HID);
}
